// GPT2Attention_5428838662886
// MI455X (gfx1250) — matrix operand order repaired and checked by reading, not compiled, not run on hardware
//
#include <hip/hip_runtime.h>

// GPT-2 attention block on gfx1250: bf16 WMMA (f32 accum) for all GEMMs,
// TDM (tensor_load_to_lds) staging of GEMM A-tiles into LDS,
// f32 softmax held entirely in LDS (CDNA5 has 320KB/WGP).

#define S_LEN 2048
#define D_DIM 1024
#define NH    16
#define HDIM  64
#define B_SZ  2
#define TOK   (B_SZ * S_LEN)   // 4096 tokens
#define D3    (3 * D_DIM)      // 3072

typedef __attribute__((ext_vector_type(16))) __bf16 v16bf;
typedef __attribute__((ext_vector_type(8)))  __bf16 v8bf;
typedef __attribute__((ext_vector_type(8)))  float  v8f;
typedef unsigned int u32x4 __attribute__((ext_vector_type(4)));
typedef int          i32x8 __attribute__((ext_vector_type(8)));
typedef int          i32x4 __attribute__((ext_vector_type(4)));

#if __has_builtin(__builtin_amdgcn_tensor_load_to_lds) && \
    __has_builtin(__builtin_amdgcn_s_wait_tensorcnt)
#define HAVE_TDM 1
#else
#define HAVE_TDM 0
#endif

__device__ __forceinline__ v8f wmma_bf16(v16bf a, v16bf b, v8f c) {
  // (neg_a, A, neg_b, B, c_mod, C, reuse_a, reuse_b)
  return __builtin_amdgcn_wmma_f32_16x16x32_bf16(false, a, false, b, (short)0, c,
                                                 false, false);
}

// ---- fragment loaders (wave32 layouts per CDNA5 ISA 7.12.2) ----
// A: 16x32 bf16. lane -> M = lane%16; elems 0..7 = K kk..kk+7, elems 8..15 = kk+16..kk+23,
//    kk = k0 + (lane<16 ? 0 : 8).  Works for global or LDS source.
__device__ __forceinline__ v16bf load_frag_a(const __bf16* base, int lda,
                                             int m0, int k0) {
  int lane = threadIdx.x & 31;
  int m  = m0 + (lane & 15);
  int kk = k0 + ((lane >> 4) << 3);
  const __bf16* p = base + (size_t)m * lda + kk;
  union { v16bf v; v8bf h[2]; } u;
  u.h[0] = *(const v8bf*)(p);
  u.h[1] = *(const v8bf*)(p + 16);
  return u.v;
}

// B: 32x16 bf16, source stored as [N][K] (row-major, ldb = K stride).
// lane -> N = lane%16; elems 0..7 = K kk..kk+7, elems 8..15 = kk+16..kk+23, kk = k0 + (lane<16 ? 0 : 8).
__device__ __forceinline__ v16bf load_frag_b(const __bf16* __restrict__ base, int ldb,
                                             int n0, int k0) {
  int lane = threadIdx.x & 31;
  int n  = n0 + (lane & 15);
  int kk = k0 + ((lane >> 4) << 3);
  const __bf16* p = base + (size_t)n * ldb + kk;
  union { v16bf v; v8bf h[2]; } u;
  u.h[0] = *(const v8bf*)(p);
  u.h[1] = *(const v8bf*)(p + 16);
  return u.v;
}

#if HAVE_TDM
// TDM: DMA a 32-row x 1024-col bf16 A-slab (row stride 1024 elems) into LDS.
// D# per CDNA5 ISA ch.8: group0 = {count, lds_addr, global_addr[56:0], type=2},
// group1 = {data_size=2B, tensor dims, tile dims, dim0 stride}.
// 6-arg builtin form (this toolchain): (g0, g1, g2, g3, extra, cpol).
__device__ __forceinline__ void tdm_load_A(const __bf16* gbase, unsigned lds_off, int m0) {
  unsigned long long ga = (unsigned long long)(uintptr_t)(gbase + (size_t)m0 * D_DIM);
  u32x4 g0 = { 1u,                                   // count=1, is_restore=0, gather off
               lds_off,                              // lds_addr (bytes)
               (unsigned)ga,                         // global_addr[31:0]
               ((unsigned)(ga >> 32) & 0x01FFFFFFu)  // global_addr[56:32]
                 | (2u << 30) };                     // type=2 ("image")
  i32x8 g1 = { (int)(1u << 16),          // workgroup_mask=0, data_size=1 (2 bytes)
               (int)(1024u << 16),       // tensor_dim0[15:0]=1024 in bits[63:48]
               (int)(32u << 16),         // tensor_dim0[31:16]=0, tensor_dim1[15:0]=32
               (int)(1024u << 16),       // tensor_dim1[31:16]=0, tile_dim0=1024
               32,                       // tile_dim1=32, tile_dim2=0
               1024,                     // tensor_dim0_stride[31:0]
               0, 0 };                   // stride hi / dim1 stride unused (2D)
  i32x4 z4 = { 0, 0, 0, 0 };
  i32x8 z8 = { 0, 0, 0, 0, 0, 0, 0, 0 };
  __builtin_amdgcn_tensor_load_to_lds(g0, g1, z4, z4, z8, 0);
}
#endif

// Cooperative fallback copy (also used by host-pass parsing): 32x1024 bf16 -> LDS
__device__ __forceinline__ void coop_load_A(const __bf16* gbase, __bf16* smem, int m0) {
  for (int c = threadIdx.x; c < 32 * 1024 / 8; c += 128) {
    int row = c >> 7;                 // (c*8)/1024
    int col = (c & 127) << 3;
    *(v8bf*)(smem + row * 1024 + col) =
        *(const v8bf*)(gbase + (size_t)(m0 + row) * D_DIM + col);
  }
}

// ---- pre-pass conversion kernels ----
__global__ void cvt_f32_bf16(const float* __restrict__ in, __bf16* __restrict__ out, int n) {
  int i = blockIdx.x * blockDim.x + threadIdx.x;
  if (i < n) out[i] = (__bf16)in[i];
}

// in[K][N] f32 -> out[N][K] bf16
__global__ void transpose_to_bf16(const float* __restrict__ in, __bf16* __restrict__ out,
                                  int K, int N) {
  int i = blockIdx.x * blockDim.x + threadIdx.x;
  if (i < K * N) {
    int k = i / N, n = i % N;
    out[(size_t)n * K + k] = (__bf16)in[i];
  }
}

// ---- QKV GEMM: [4096,1024] x [1024,3072] + bias, routed into head layouts ----
// grid: 128 M32-tiles * 12 N256-tiles; block = 128 (4 waves).
// Block stages its 32x1024 A-slab in LDS via TDM; wave w owns cols n256*256+w*64..+63,
// computing a 32x64 tile (8 accumulators, 2 A-frags x 4 B-frags per K-step).
__global__ __launch_bounds__(128) void qkv_gemm(const __bf16* __restrict__ xb,
                                                const __bf16* __restrict__ wt,   // [3072][1024]
                                                const float*  __restrict__ bias, // [3072]
                                                __bf16* __restrict__ qb,   // [B,H,S,64], pre-scaled
                                                __bf16* __restrict__ kb,   // [B,H,S,64]
                                                __bf16* __restrict__ vtb)  // [B,H,64,S]
{
  __shared__ __align__(16) __bf16 Atile[32 * 1024];   // 64 KB
  int n256  = blockIdx.x % 12;
  int mtile = blockIdx.x / 12;
  int wave  = threadIdx.x >> 5;
  int lane  = threadIdx.x & 31;
  int m0  = mtile * 32;
  int n0w = n256 * 256 + wave * 64;

#if HAVE_TDM
  if (wave == 0) {
    tdm_load_A(xb, (unsigned)(uintptr_t)(void*)Atile, m0);
    __builtin_amdgcn_s_wait_tensorcnt(0);
  }
#else
  coop_load_A(xb, Atile, m0);
#endif
  __syncthreads();

  v8f acc[2][4] = {};
  for (int k0 = 0; k0 < D_DIM; k0 += 32) {
    __builtin_prefetch(wt + (size_t)n0w * D_DIM + k0 + 256, 0, 1);
    v16bf a0 = load_frag_a(Atile, D_DIM, 0, k0);
    v16bf a1 = load_frag_a(Atile, D_DIM, 16, k0);
#pragma unroll
    for (int t = 0; t < 4; ++t) {
      v16bf b = load_frag_b(wt, D_DIM, n0w + t * 16, k0);
      acc[0][t] = wmma_bf16(a0, b, acc[0][t]);
      acc[1][t] = wmma_bf16(a1, b, acc[1][t]);
    }
  }
#pragma unroll
  for (int t = 0; t < 4; ++t) {
    int n  = n0w + t * 16 + (lane & 15);
    float bv = bias[n];
#pragma unroll
    for (int mi = 0; mi < 2; ++mi) {
#pragma unroll
      for (int r = 0; r < 8; ++r) {
        int m = m0 + mi * 16 + r + ((lane >> 4) << 3);
        float val = acc[mi][t][r] + bv;
        int b_ = m >> 11;        // token / 2048
        int s_ = m & 2047;
        if (n < D_DIM) {                       // Q (fold 1/sqrt(64))
          int h = n >> 6, hd = n & 63;
          qb[(((size_t)(b_ * NH + h) * S_LEN) + s_) * HDIM + hd] = (__bf16)(val * 0.125f);
        } else if (n < 2 * D_DIM) {            // K
          int nn = n - D_DIM; int h = nn >> 6, hd = nn & 63;
          kb[(((size_t)(b_ * NH + h) * S_LEN) + s_) * HDIM + hd] = (__bf16)val;
        } else {                               // V, transposed
          int nn = n - 2 * D_DIM; int h = nn >> 6, hd = nn & 63;
          vtb[(((size_t)(b_ * NH + h) * HDIM) + hd) * S_LEN + s_] = (__bf16)val;
        }
      }
    }
  }
}

// ---- causal attention: one block per (b,h,16-query tile); 4 waves ----
__global__ __launch_bounds__(128) void attn_kernel(const __bf16* __restrict__ qb,
                                                   const __bf16* __restrict__ kbm,
                                                   const __bf16* __restrict__ vtb,
                                                   __bf16* __restrict__ aout) // [B*S, D] bf16
{
  __shared__ float sc[16][S_LEN];     // full score stripe for 16 queries (128 KB)
  __shared__ float red[16][8];
  __shared__ float rowstat[16];

  int qtile = blockIdx.x & 127;       // S/16
  int bh    = blockIdx.x >> 7;        // b*16+h
  int q0    = qtile * 16;
  int kmax  = q0 + 16;                // causal: keys 0..q
  int wave  = threadIdx.x >> 5;
  int lane  = threadIdx.x & 31;

  const __bf16* Q  = qb  + (size_t)bh * S_LEN * HDIM;
  const __bf16* K  = kbm + (size_t)bh * S_LEN * HDIM;
  const __bf16* Vt = vtb + (size_t)bh * HDIM * S_LEN;

  // phase A: scores = (Q*0.125) @ K^T  (wave-uniform tile loop, EXEC stays full)
  int ntiles = kmax >> 4;
  for (int nt = wave; nt < ntiles; nt += 4) {
    int n0 = nt * 16;
    v8f acc = {};
#pragma unroll
    for (int k0 = 0; k0 < HDIM; k0 += 32) {
      v16bf a = load_frag_a(Q, HDIM, q0, k0);
      v16bf b = load_frag_b(K, HDIM, n0, k0);   // K rows are K^T columns
      acc = wmma_bf16(a, b, acc);
    }
    int n = n0 + (lane & 15);
#pragma unroll
    for (int r = 0; r < 8; ++r) {
      int qrow = r + ((lane >> 4) << 3);
      int qg   = q0 + qrow;
      sc[qrow][n] = (n > qg) ? -1e30f : acc[r];
    }
  }
  __syncthreads();

  // phase B/C: fp32 softmax stats (128 threads = 16 rows x 8 cols-strided)
  int r = threadIdx.x >> 3, j = threadIdx.x & 7;
  float mx = -1e30f;
  for (int c = j; c < kmax; c += 8) mx = fmaxf(mx, sc[r][c]);
  red[r][j] = mx;
  __syncthreads();
  if (j == 0) {
    float m2 = red[r][0];
#pragma unroll
    for (int t = 1; t < 8; ++t) m2 = fmaxf(m2, red[r][t]);
    rowstat[r] = m2;
  }
  __syncthreads();
  float rmax = rowstat[r];
  int kpad = (kmax + 31) & ~31;       // pad to WMMA K step, zero-filled
  float sum = 0.f;
  for (int c = j; c < kpad; c += 8) {
    float e = 0.f;
    if (c < kmax) e = __expf(sc[r][c] - rmax);
    sc[r][c] = e;                     // unnormalized P
    sum += e;
  }
  red[r][j] = sum;
  __syncthreads();
  if (j == 0) {
    float s2 = 0.f;
#pragma unroll
    for (int t = 0; t < 8; ++t) s2 += red[r][t];
    rowstat[r] = s2;                  // row sum (normalize in epilogue)
  }
  __syncthreads();

  // phase D: O = P @ V ; wave w owns hd columns w*16..w*16+15
  int n0v = wave * 16;
  v8f acc = {};
  for (int kb2 = 0; kb2 < kpad; kb2 += 32) {
    int am = lane & 15;
    int kk = kb2 + ((lane >> 4) << 3);
    union { v16bf v; __bf16 e[16]; } ua;
#pragma unroll
    for (int e2 = 0; e2 < 8; ++e2) {
      ua.e[e2]     = (__bf16)sc[am][kk + e2];
      ua.e[e2 + 8] = (__bf16)sc[am][kk + 16 + e2];
    }
    v16bf b = load_frag_b(Vt, S_LEN, n0v, kb2);  // Vt rows = V columns
    acc = wmma_bf16(ua.v, b, acc);
  }
  int hd = n0v + (lane & 15);
  int b_ = bh >> 4, h = bh & 15;
#pragma unroll
  for (int rr = 0; rr < 8; ++rr) {
    int qrow = rr + ((lane >> 4) << 3);
    float inv = 1.0f / rowstat[qrow];
    int s_ = q0 + qrow;
    aout[((size_t)(b_ * S_LEN + s_)) * D_DIM + h * HDIM + hd] = (__bf16)(acc[rr] * inv);
  }
}

// ---- proj GEMM: [4096,1024] x [1024,1024] + bias -> f32 out ----
// grid: 128 M32-tiles * 4 N256-tiles; same TDM-staged structure as qkv_gemm.
__global__ __launch_bounds__(128) void proj_gemm(const __bf16* __restrict__ ab,
                                                 const __bf16* __restrict__ wt,   // [1024][1024]
                                                 const float*  __restrict__ bias,
                                                 float* __restrict__ out) {
  __shared__ __align__(16) __bf16 Atile[32 * 1024];   // 64 KB
  int n256  = blockIdx.x & 3;
  int mtile = blockIdx.x >> 2;
  int wave  = threadIdx.x >> 5;
  int lane  = threadIdx.x & 31;
  int m0  = mtile * 32;
  int n0w = n256 * 256 + wave * 64;

#if HAVE_TDM
  if (wave == 0) {
    tdm_load_A(ab, (unsigned)(uintptr_t)(void*)Atile, m0);
    __builtin_amdgcn_s_wait_tensorcnt(0);
  }
#else
  coop_load_A(ab, Atile, m0);
#endif
  __syncthreads();

  v8f acc[2][4] = {};
  for (int k0 = 0; k0 < D_DIM; k0 += 32) {
    __builtin_prefetch(wt + (size_t)n0w * D_DIM + k0 + 256, 0, 1);
    v16bf a0 = load_frag_a(Atile, D_DIM, 0, k0);
    v16bf a1 = load_frag_a(Atile, D_DIM, 16, k0);
#pragma unroll
    for (int t = 0; t < 4; ++t) {
      v16bf b = load_frag_b(wt, D_DIM, n0w + t * 16, k0);
      acc[0][t] = wmma_bf16(a0, b, acc[0][t]);
      acc[1][t] = wmma_bf16(a1, b, acc[1][t]);
    }
  }
#pragma unroll
  for (int t = 0; t < 4; ++t) {
    int n = n0w + t * 16 + (lane & 15);
    float bv = bias[n];
#pragma unroll
    for (int mi = 0; mi < 2; ++mi) {
#pragma unroll
      for (int r = 0; r < 8; ++r) {
        int m = m0 + mi * 16 + r + ((lane >> 4) << 3);
        out[(size_t)m * D_DIM + n] = acc[mi][t][r] + bv;
      }
    }
  }
}

extern "C" void kernel_launch(void* const* d_in, const int* in_sizes, int n_in,
                              void* d_out, int out_size, void* d_ws, size_t ws_size,
                              hipStream_t stream) {
  const float* x      = (const float*)d_in[0];   // [2,2048,1024]
  const float* w_attn = (const float*)d_in[1];   // [1024,3072]
  const float* b_attn = (const float*)d_in[2];   // [3072]
  const float* w_proj = (const float*)d_in[3];   // [1024,1024]
  const float* b_proj = (const float*)d_in[4];   // [1024]
  float* out = (float*)d_out;

  char* ws = (char*)d_ws;
  __bf16* xb    = (__bf16*)(ws);                         //  8 MB  x bf16 [4096,1024]
  __bf16* watT  = (__bf16*)(ws + (size_t)8  * 1048576);  //  6 MB  w_attn^T bf16 [3072,1024]
  __bf16* wprT  = (__bf16*)(ws + (size_t)14 * 1048576);  //  2 MB  w_proj^T bf16 [1024,1024]
  __bf16* qb    = (__bf16*)(ws + (size_t)16 * 1048576);  //  8 MB  Q bf16 [B,H,S,64]
  __bf16* kb    = (__bf16*)(ws + (size_t)24 * 1048576);  //  8 MB  K bf16 [B,H,S,64]
  __bf16* vtb   = (__bf16*)(ws + (size_t)32 * 1048576);  //  8 MB  V^T bf16 [B,H,64,S]
  __bf16* aoutb = (__bf16*)(ws + (size_t)40 * 1048576);  //  8 MB  attn-out bf16 [4096,1024]

  // pre-pass conversions
  {
    int n = TOK * D_DIM;
    cvt_f32_bf16<<<(n + 255) / 256, 256, 0, stream>>>(x, xb, n);
  }
  {
    int n = D_DIM * D3;
    transpose_to_bf16<<<(n + 255) / 256, 256, 0, stream>>>(w_attn, watT, D_DIM, D3);
  }
  {
    int n = D_DIM * D_DIM;
    transpose_to_bf16<<<(n + 255) / 256, 256, 0, stream>>>(w_proj, wprT, D_DIM, D_DIM);
  }

  // QKV GEMM: 128 M32-tiles x 12 N256-tiles
  qkv_gemm<<<128 * 12, 128, 0, stream>>>(xb, watT, b_attn, qb, kb, vtb);

  // attention: (B*H) * (S/16) blocks
  attn_kernel<<<B_SZ * NH * (S_LEN / 16), 128, 0, stream>>>(qb, kb, vtb, aoutb);

  // proj GEMM: 128 M32-tiles x 4 N256-tiles
  proj_gemm<<<128 * 4, 128, 0, stream>>>(aoutb, wprT, b_proj, out);
}
